// RNNModel_1889785610863
// MI455X (gfx1250) — compile-verified, matrix operand order checked by reading, not run on hardware
//
#include <hip/hip_runtime.h>

#define TLEN  1024
#define BATCH 512
#define EMB   64
#define HID   64
#define G4    256   // 4*HID
#define RB    16    // batch rows per block
#define STR   72    // padded LDS row stride (bf16 elems, 144B, 16B-aligned, conflict-free)

typedef __attribute__((ext_vector_type(16))) __bf16 v16bf;
typedef __attribute__((ext_vector_type(8)))  float  v8f;

union ABop  { v16bf v; uint4 q[2]; };
union Mask8 { uint4 q[2]; int m[8]; };

__device__ __forceinline__ unsigned short f2bf(float f) {
    unsigned u = __float_as_uint(f);
    u += 0x7FFFu + ((u >> 16) & 1u);      // round-to-nearest-even
    return (unsigned short)(u >> 16);
}
__device__ __forceinline__ float sigm_(float x) {
    return 1.0f / (1.0f + __expf(-x));
}
__device__ __forceinline__ float tanh_(float x) {
    float t = __expf(-2.0f * x);
    return (1.0f - t) / (1.0f + t);
}

// One block = 16 batch rows, full T recurrence, one direction (blockIdx.y).
// 4 waves; wave w owns N-columns [16w,16w+16) of each of the 4 gates.
// W/U operand tiles live in registers for the whole T loop; sX/sH/sMask are
// ping-pong double-buffered so a single barrier per timestep suffices.
__global__ __launch_bounds__(128, 1)
void lstm_kernel(const int* __restrict__ x, const float* __restrict__ emb,
                 const float* __restrict__ Wf, const float* __restrict__ Uf,
                 const float* __restrict__ bfv,
                 const float* __restrict__ Wb, const float* __restrict__ Ub,
                 const float* __restrict__ bbv,
                 float* __restrict__ hout) {
    __shared__ __align__(16) unsigned short sW[G4 * STR];      // W^T[n][k] bf16 (init only)
    __shared__ __align__(16) unsigned short sU[G4 * STR];      // U^T[n][k] bf16 (init only)
    __shared__ __align__(16) unsigned short sX[2][RB * STR];   // x_t tile  bf16 (ping-pong)
    __shared__ __align__(16) unsigned short sH[2][RB * STR];   // h state   bf16 (ping-pong)
    __shared__ __align__(16) int sMask[2][RB];
    __shared__ float sB[G4];

    const int tid   = threadIdx.x;
    const int dir   = blockIdx.y;
    const int brow0 = blockIdx.x * RB;

    const float* W  = dir ? Wb  : Wf;
    const float* U  = dir ? Ub  : Uf;
    const float* bv = dir ? bbv : bfv;

    // Stage weights transposed into LDS as bf16 (coalesced global reads).
    for (int i = tid; i < EMB * G4; i += 128) {
        int k = i >> 8;       // row of W (EMB)
        int n = i & 255;      // col of W (4H)
        sW[n * STR + k] = f2bf(W[i]);
        sU[n * STR + k] = f2bf(U[i]);
    }
    for (int i = tid; i < G4; i += 128) sB[i] = bv[i];
    for (int i = tid; i < RB * STR; i += 128) sH[0][i] = 0;

    const int lane  = tid & 31;
    const int wv    = tid >> 5;    // wave 0..3
    const int hh    = lane >> 4;   // lane half
    const int nl    = lane & 15;
    const int koffA = hh * 8;      // K sub-offset per ISA 16-bit operand layout

    const int gr = tid >> 3;       // gather: row 0..15
    const int gp = tid & 7;        // gather: 8-float chunk 0..7

    // gather for timestep t into ping-pong buffer `buf`
    auto do_gather = [&](int t, int buf) {
        const int idx = x[(brow0 + gr) * TLEN + t];
        if (gp == 0) sMask[buf][gr] = (idx != 0);
        const float* e = emb + idx * EMB + gp * 8;
        float4 e0 = *(const float4*)(e);
        float4 e1 = *(const float4*)(e + 4);
        uint4 pk;
        pk.x = (unsigned)f2bf(e0.x) | ((unsigned)f2bf(e0.y) << 16);
        pk.y = (unsigned)f2bf(e0.z) | ((unsigned)f2bf(e0.w) << 16);
        pk.z = (unsigned)f2bf(e1.x) | ((unsigned)f2bf(e1.y) << 16);
        pk.w = (unsigned)f2bf(e1.z) | ((unsigned)f2bf(e1.w) << 16);
        *(uint4*)&sX[buf][gr * STR + gp * 8] = pk;
    };

    __syncthreads();   // weights staged

    // Hoist all B operands (this wave's 4 gate columns, W and U, 2 K-chunks)
    // into registers for the entire recurrence: 16 operands = 128 VGPRs.
    ABop bW[4][2], bU[4][2];
    float biasv[4];
#pragma unroll
    for (int g = 0; g < 4; ++g) {
        const int col = g * HID + wv * 16 + nl;
        biasv[g] = sB[col];
        const unsigned short* wr = &sW[col * STR];
        const unsigned short* ur = &sU[col * STR];
        bW[g][0].q[0] = *(const uint4*)(wr + 0  + koffA);
        bW[g][0].q[1] = *(const uint4*)(wr + 16 + koffA);
        bW[g][1].q[0] = *(const uint4*)(wr + 32 + koffA);
        bW[g][1].q[1] = *(const uint4*)(wr + 48 + koffA);
        bU[g][0].q[0] = *(const uint4*)(ur + 0  + koffA);
        bU[g][0].q[1] = *(const uint4*)(ur + 16 + koffA);
        bU[g][1].q[0] = *(const uint4*)(ur + 32 + koffA);
        bU[g][1].q[1] = *(const uint4*)(ur + 48 + koffA);
    }

    float creg[8], hreg[8];
#pragma unroll
    for (int v = 0; v < 8; ++v) { creg[v] = 0.0f; hreg[v] = 0.0f; }

    // prime the pipeline: gather first timestep into buffer 0
    do_gather(dir ? (TLEN - 1) : 0, 0);
    __syncthreads();

    for (int tt = 0; tt < TLEN; ++tt) {
        const int pb = tt & 1;          // read set
        const int nb = pb ^ 1;          // write set

        // ---- issue next timestep's gather early (hides global latency) ----
        if (tt + 1 < TLEN) {
            const int tn = dir ? (TLEN - 2 - tt) : (tt + 1);
            do_gather(tn, nb);
        }

        // ---- A operands for this wave ----
        ABop aX0, aX1, aH0, aH1;
        {
            const unsigned short* xr = &sX[pb][nl * STR];
            const unsigned short* hr = &sH[pb][nl * STR];
            aX0.q[0] = *(const uint4*)(xr + 0  + koffA);
            aX0.q[1] = *(const uint4*)(xr + 16 + koffA);
            aX1.q[0] = *(const uint4*)(xr + 32 + koffA);
            aX1.q[1] = *(const uint4*)(xr + 48 + koffA);
            aH0.q[0] = *(const uint4*)(hr + 0  + koffA);
            aH0.q[1] = *(const uint4*)(hr + 16 + koffA);
            aH1.q[0] = *(const uint4*)(hr + 32 + koffA);
            aH1.q[1] = *(const uint4*)(hr + 48 + koffA);
        }

        // ---- z = x@W + h@U for this wave's 16 columns of each gate ----
        // First WMMA of each chain starts from inline C=0; bias added later.
        v8f z[4];
#pragma unroll
        for (int g = 0; g < 4; ++g) {
            v8f acc = {};
            acc = __builtin_amdgcn_wmma_f32_16x16x32_bf16(false, aX0.v, false, bW[g][0].v,
                                                          (short)0, acc, false, false);
            acc = __builtin_amdgcn_wmma_f32_16x16x32_bf16(false, aX1.v, false, bW[g][1].v,
                                                          (short)0, acc, false, false);
            acc = __builtin_amdgcn_wmma_f32_16x16x32_bf16(false, aH0.v, false, bU[g][0].v,
                                                          (short)0, acc, false, false);
            acc = __builtin_amdgcn_wmma_f32_16x16x32_bf16(false, aH1.v, false, bU[g][1].v,
                                                          (short)0, acc, false, false);
            z[g] = acc;
        }

        // ---- masks for this lane's 8 rows (two b128 LDS loads) ----
        Mask8 mk;
        mk.q[0] = *(const uint4*)&sMask[pb][hh * 8];
        mk.q[1] = *(const uint4*)&sMask[pb][hh * 8 + 4];

        // ---- gate math; element v == (row v+8*hh, hidden col wv*16+nl) ----
#pragma unroll
        for (int v = 0; v < 8; ++v) {
            float iv = sigm_(z[0][v] + biasv[0]);
            float fv = sigm_(z[1][v] + biasv[1]);
            float gv = tanh_(z[2][v] + biasv[2]);
            float ov = sigm_(z[3][v] + biasv[3]);
            float cn = fv * creg[v] + iv * gv;
            float hn = ov * tanh_(cn);
            const bool m = mk.m[v] != 0;
            creg[v] = m ? cn : creg[v];
            hreg[v] = m ? hn : hreg[v];
        }

        // ---- publish h_t into the other buffer; one barrier per step ----
#pragma unroll
        for (int v = 0; v < 8; ++v)
            sH[nb][(v + hh * 8) * STR + wv * 16 + nl] = f2bf(hreg[v]);

        __syncthreads();
    }

    // final h for this direction -> workspace (f32)
#pragma unroll
    for (int v = 0; v < 8; ++v) {
        const int b = brow0 + v + hh * 8;
        const int j = wv * 16 + nl;
        hout[(dir * BATCH + b) * HID + j] = hreg[v];
    }
}

// Head: out = relu(concat(h_fwd,h_bwd) @ W1 + b1) @ W2 + b2   (512 rows)
__global__ __launch_bounds__(128)
void head_kernel(const float* __restrict__ hbuf,
                 const float* __restrict__ W1, const float* __restrict__ b1,
                 const float* __restrict__ W2, const float* __restrict__ b2,
                 float* __restrict__ out) {
    __shared__ float sW1[128 * 32];
    __shared__ float sW2[32];
    __shared__ float sB1[32];
    const int tid = threadIdx.x;
    for (int i = tid; i < 128 * 32; i += 128) sW1[i] = W1[i];
    if (tid < 32) { sW2[tid] = W2[tid]; sB1[tid] = b1[tid]; }
    __syncthreads();

    const int row = blockIdx.x * 128 + tid;
    const float* hf = hbuf + row * HID;
    const float* hb = hbuf + BATCH * HID + row * HID;

    float acc[32];
#pragma unroll
    for (int k = 0; k < 32; ++k) acc[k] = sB1[k];
    for (int j = 0; j < 64; ++j) {
        const float hv = hf[j];
#pragma unroll
        for (int k = 0; k < 32; ++k) acc[k] += hv * sW1[j * 32 + k];
    }
    for (int j = 0; j < 64; ++j) {
        const float hv = hb[j];
#pragma unroll
        for (int k = 0; k < 32; ++k) acc[k] += hv * sW1[(64 + j) * 32 + k];
    }
    float s = b2[0];
#pragma unroll
    for (int k = 0; k < 32; ++k) s += fmaxf(acc[k], 0.0f) * sW2[k];
    out[row] = s;
}

extern "C" void kernel_launch(void* const* d_in, const int* in_sizes, int n_in,
                              void* d_out, int out_size, void* d_ws, size_t ws_size,
                              hipStream_t stream) {
    const int*   x   = (const int*)d_in[0];
    const float* emb = (const float*)d_in[1];
    const float* Wf  = (const float*)d_in[2];
    const float* Uf  = (const float*)d_in[3];
    const float* bf_ = (const float*)d_in[4];
    const float* Wb  = (const float*)d_in[5];
    const float* Ub  = (const float*)d_in[6];
    const float* bb_ = (const float*)d_in[7];
    const float* W1  = (const float*)d_in[8];
    const float* b1  = (const float*)d_in[9];
    const float* W2  = (const float*)d_in[10];
    const float* b2  = (const float*)d_in[11];

    float* hbuf = (float*)d_ws;           // [2][512][64] f32 final hidden states
    float* out  = (float*)d_out;          // [512][1] f32

    lstm_kernel<<<dim3(BATCH / RB, 2), 128, 0, stream>>>(
        x, emb, Wf, Uf, bf_, Wb, Ub, bb_, hbuf);
    head_kernel<<<dim3(BATCH / 128), 128, 0, stream>>>(
        hbuf, W1, b1, W2, b2, out);
}
